// GASADecoder_26173530701973
// MI455X (gfx1250) — matrix operand order repaired and checked by reading, not compiled, not run on hardware
//
#include <hip/hip_runtime.h>
#include <hip/hip_bf16.h>

typedef __attribute__((ext_vector_type(16))) _Float16 v16h;
typedef __attribute__((ext_vector_type(8)))  _Float16 h8;
typedef __attribute__((ext_vector_type(8)))  float    v8f;
typedef __attribute__((ext_vector_type(4)))  float    f4;

static __device__ __forceinline__ v8f wmma_f16(v16h a, v16h b, v8f c) {
    // D = A(16x32 f16) * B(32x16 f16) + C(16x16 f32)
    return __builtin_amdgcn_wmma_f32_16x16x32_f16(false, a, false, b, (short)0, c, false, false);
}

// ---- fragment loaders (16-bit WMMA operands, contiguous k order in A and B) -------

// A fragment 16x32: lane m=l&15, g=l>>4; k = 16g .. 16g+15 (contiguous)
// base = &A[row0][kk] (row-major f32, ld in floats); optional scale folded in.
static __device__ __forceinline__ v16h load_a_frag(const float* __restrict__ base,
                                                   int ld, int lane, float scale) {
    const int m = lane & 15, g = lane >> 4;
    const float* p = base + (size_t)m * ld + 16 * g;
    f4 c0 = *(const f4*)(p);
    f4 c1 = *(const f4*)(p + 4);
    f4 c2 = *(const f4*)(p + 8);
    f4 c3 = *(const f4*)(p + 12);
    v16h r;
#pragma unroll
    for (int i = 0; i < 4; ++i) {
        r[i]      = (_Float16)(c0[i] * scale);
        r[4 + i]  = (_Float16)(c1[i] * scale);
        r[8 + i]  = (_Float16)(c2[i] * scale);
        r[12 + i] = (_Float16)(c3[i] * scale);
    }
    return r;
}

// B fragment 32x16 for A*B^T: W stored row-major [N,K]; lane n=l&15, g=l>>4,
// holds contiguous k = 16g .. 16g+15.  base = &W[n0][kk].
static __device__ __forceinline__ v16h load_b_frag_nt(const float* __restrict__ base,
                                                      int ld, int lane) {
    const int n = lane & 15, g = lane >> 4;
    const float* p = base + (size_t)n * ld + 16 * g;
    f4 c0 = *(const f4*)(p);
    f4 c1 = *(const f4*)(p + 4);
    f4 c2 = *(const f4*)(p + 8);
    f4 c3 = *(const f4*)(p + 12);
    v16h r;
#pragma unroll
    for (int i = 0; i < 4; ++i) {
        r[i]      = (_Float16)c0[i];
        r[4 + i]  = (_Float16)c1[i];
        r[8 + i]  = (_Float16)c2[i];
        r[12 + i] = (_Float16)c3[i];
    }
    return r;
}

// B fragment 32x16 for A*B (NN): B row-major [K,N]; element t -> B[k0+t+16g][n0+n]
static __device__ __forceinline__ v16h load_b_frag_nn(const float* __restrict__ base,
                                                      int ld, int lane) {
    const int n = lane & 15, g = lane >> 4;
    const float* p = base + (size_t)(16 * g) * ld + n;
    v16h r;
#pragma unroll
    for (int t = 0; t < 16; ++t) r[t] = (_Float16)p[(size_t)t * ld];
    return r;
}

// P fragment (16x32) from LDS row-major f16 tile
static __device__ __forceinline__ v16h load_p_frag_lds(const _Float16* sP, int lane) {
    const int m = lane & 15, g = lane >> 4;
    h8 a = *(const h8*)(sP + m * 32 + 16 * g);
    h8 b = *(const h8*)(sP + m * 32 + 16 * g + 8);
    v16h r;
#pragma unroll
    for (int i = 0; i < 8; ++i) { r[i] = a[i]; r[8 + i] = b[i]; }
    return r;
}

static __device__ __forceinline__ void store_tile(v8f acc, float* __restrict__ C,
                                                  const float* __restrict__ bias,
                                                  int n, int m0, int ldc, int relu) {
    const float bv = bias ? bias[n] : 0.0f;
#pragma unroll
    for (int r = 0; r < 8; ++r) {
        float v = acc[r] + bv;
        if (relu) v = fmaxf(v, 0.0f);
        C[(size_t)(m0 + r) * ldc + n] = v;
    }
}

// ---- generic NT GEMM: C[M,N] = act(A[M,K] * W[N,K]^T + bias[N]) ----------
// block = 128 threads (4 waves). Each wave owns a 16x64 output strip
// (4 accumulators): one A fragment is reused across 4 B fragments per
// k-step -> 4 independent WMMAs to hide VMEM/cvt latency + hazard NOPs.
// grid(N/64, M/64).
__global__ __launch_bounds__(128) void gemm_nt_kernel(
    const float* __restrict__ A, const float* __restrict__ W,
    const float* __restrict__ bias, float* __restrict__ C,
    int K, int lda, int ldb, int ldc, int relu)
{
    const int lane  = threadIdx.x & 31;
    const int wave  = threadIdx.x >> 5;
    const int tileN = blockIdx.x * 64;
    const int tileM = blockIdx.y * 64 + wave * 16;
    const float* Abase = A + (size_t)tileM * lda;
    const float* B0 = W + (size_t)(tileN)      * ldb;
    const float* B1 = W + (size_t)(tileN + 16) * ldb;
    const float* B2 = W + (size_t)(tileN + 32) * ldb;
    const float* B3 = W + (size_t)(tileN + 48) * ldb;
    v8f a0 = {}, a1 = {}, a2 = {}, a3 = {};
    for (int kk = 0; kk < K; kk += 32) {
        v16h af = load_a_frag(Abase + kk, lda, lane, 1.0f);
        v16h b0 = load_b_frag_nt(B0 + kk, ldb, lane);
        v16h b1 = load_b_frag_nt(B1 + kk, ldb, lane);
        v16h b2 = load_b_frag_nt(B2 + kk, ldb, lane);
        v16h b3 = load_b_frag_nt(B3 + kk, ldb, lane);
        a0 = wmma_f16(af, b0, a0);
        a1 = wmma_f16(af, b1, a1);
        a2 = wmma_f16(af, b2, a2);
        a3 = wmma_f16(af, b3, a3);
    }
    const int n  = lane & 15;
    const int m0 = tileM + 8 * (lane >> 4);
    store_tile(a0, C, bias, tileN + n,      m0, ldc, relu);
    store_tile(a1, C, bias, tileN + 16 + n, m0, ldc, relu);
    store_tile(a2, C, bias, tileN + 32 + n, m0, ldc, relu);
    store_tile(a3, C, bias, tileN + 48 + n, m0, ldc, relu);
}

// ---- streaming attention: one wave per (b, h, 16-query tile) -------------
// O[b*Qn+q, h*64 + :] = softmax(scale*Q K^T + beta*bias) V   (online softmax)
__global__ __launch_bounds__(32) void attn_kernel(
    const float* __restrict__ Qm, const float* __restrict__ Km,
    const float* __restrict__ Vm, const float* __restrict__ biasB,
    const float* __restrict__ betaPtr, float* __restrict__ Om,
    int Qn, int Lk, int ldq, int ldk, int ldo, float scale)
{
    __shared__ __align__(16) float    sS[16 * 32];
    __shared__ __align__(16) _Float16 sP[16 * 32];
    __shared__ float sM[16], sL[16], sC[16];

    const int lane = threadIdx.x;
    const int qt = blockIdx.x, h = blockIdx.y, b = blockIdx.z;
    const int qrow0 = b * Qn + qt * 16;
    const int krow0 = b * Lk;
    const int ho = h * 64;
    const float beta = biasB ? betaPtr[0] : 0.0f;

    // preload Q fragments (HD=64 -> two 16x32 k-chunks), scale folded in
    const float* Qbase = Qm + (size_t)qrow0 * ldq + ho;
    const v16h qf0 = load_a_frag(Qbase,      ldq, lane, scale);
    const v16h qf1 = load_a_frag(Qbase + 32, ldq, lane, scale);

    if (lane < 16) { sM[lane] = -3.0e38f; sL[lane] = 0.0f; }
    v8f o0 = {}, o1 = {}, o2 = {}, o3 = {};
    __syncthreads();

    for (int k0 = 0; k0 < Lk; k0 += 32) {
        // S tile 16x32 = Q * K^T  (two 16-key halves)
        const float* Kb0 = Km + (size_t)(krow0 + k0) * ldk + ho;
        const float* Kb1 = Km + (size_t)(krow0 + k0 + 16) * ldk + ho;
        v8f s0 = {}, s1 = {};
        s0 = wmma_f16(qf0, load_b_frag_nt(Kb0,      ldk, lane), s0);
        s0 = wmma_f16(qf1, load_b_frag_nt(Kb0 + 32, ldk, lane), s0);
        s1 = wmma_f16(qf0, load_b_frag_nt(Kb1,      ldk, lane), s1);
        s1 = wmma_f16(qf1, load_b_frag_nt(Kb1 + 32, ldk, lane), s1);
        {   // spill accumulator-layout S into LDS row-major
            const int n = lane & 15, g = lane >> 4;
#pragma unroll
            for (int r = 0; r < 8; ++r) {
                sS[(r + 8 * g) * 32 + n]      = s0[r];
                sS[(r + 8 * g) * 32 + 16 + n] = s1[r];
            }
        }
        __syncthreads();
        // per-row online softmax (lane m owns row m)
        if (lane < 16) {
            const int m = lane;
            const float mo = sM[m], lo = sL[m];
            const float* bp = biasB ? (biasB + (size_t)(qrow0 + m) * Lk + k0) : nullptr;
            float sv[32];
            float mx = mo;
#pragma unroll
            for (int j = 0; j < 32; ++j) {
                float v = sS[m * 32 + j];
                if (bp) v += beta * bp[j];
                sv[j] = v;
                mx = fmaxf(mx, v);
            }
            const float corr = __expf(mo - mx);
            float sum = lo * corr;
#pragma unroll
            for (int j = 0; j < 32; ++j) {
                float p = __expf(sv[j] - mx);
                sum += p;
                sP[m * 32 + j] = (_Float16)p;
            }
            sM[m] = mx; sL[m] = sum; sC[m] = corr;
        }
        __syncthreads();
        {   // rescale running O by per-row correction factor
            const int g = lane >> 4;
#pragma unroll
            for (int r = 0; r < 8; ++r) {
                const float c = sC[r + 8 * g];
                o0[r] *= c; o1[r] *= c; o2[r] *= c; o3[r] *= c;
            }
        }
        // O += P(16x32) * V(32x64)
        const v16h pf = load_p_frag_lds(sP, lane);
        const float* Vb = Vm + (size_t)(krow0 + k0) * ldk + ho;
        o0 = wmma_f16(pf, load_b_frag_nn(Vb,      ldk, lane), o0);
        o1 = wmma_f16(pf, load_b_frag_nn(Vb + 16, ldk, lane), o1);
        o2 = wmma_f16(pf, load_b_frag_nn(Vb + 32, ldk, lane), o2);
        o3 = wmma_f16(pf, load_b_frag_nn(Vb + 48, ldk, lane), o3);
        __syncthreads();
    }
    // normalize and store
    {
        const int n = lane & 15, g = lane >> 4;
#pragma unroll
        for (int r = 0; r < 8; ++r) {
            const float inv = 1.0f / sL[r + 8 * g];
            float* op = Om + (size_t)(qrow0 + r + 8 * g) * ldo + ho + n;
            op[0]  = o0[r] * inv;
            op[16] = o1[r] * inv;
            op[32] = o2[r] * inv;
            op[48] = o3[r] * inv;
        }
    }
}

// ---- distance-kernel bias: bias[b,q,l] = clip(MLP(|qp-mp|), -10, 0) ------
__global__ __launch_bounds__(256) void distbias_kernel(
    const float* __restrict__ qp, const float* __restrict__ mp,
    const float* __restrict__ w1, const float* __restrict__ b1,
    const float* __restrict__ w2, const float* __restrict__ b2,
    float* __restrict__ out, int Qn, int Lk, long total)
{
    __shared__ float sw1[32], sb1[32], sw2[32];
    if (threadIdx.x < 32) {
        sw1[threadIdx.x] = w1[threadIdx.x];
        sb1[threadIdx.x] = b1[threadIdx.x];
        sw2[threadIdx.x] = w2[threadIdx.x];
    }
    __syncthreads();
    const long idx = (long)blockIdx.x * blockDim.x + threadIdx.x;
    if (idx >= total) return;
    const int  l = (int)(idx % Lk);
    const long t = idx / Lk;
    const int  q = (int)(t % Qn);
    const int  b = (int)(t / Qn);
    const float* qpp = qp + (size_t)(b * Qn + q) * 3;
    const float* mpp = mp + (size_t)(b * Lk + l) * 3;
    const float dx = qpp[0] - mpp[0], dy = qpp[1] - mpp[1], dz = qpp[2] - mpp[2];
    const float dist = sqrtf(fmaxf(dx * dx + dy * dy + dz * dz, 1e-12f));
    float acc = b2[0];
#pragma unroll
    for (int i = 0; i < 32; ++i) {
        const float hv = fmaxf(dist * sw1[i] + sb1[i], 0.0f);
        acc += hv * sw2[i];
    }
    out[idx] = fminf(fmaxf(acc, -10.0f), 0.0f);
}

// ---- out = LayerNorm(a + r) * g + b ; one block per row, D=512 -----------
__global__ __launch_bounds__(256) void add_ln_kernel(
    const float* __restrict__ a, const float* __restrict__ r,
    const float* __restrict__ g, const float* __restrict__ be,
    float* __restrict__ out)
{
    constexpr int D = 512;
    __shared__ float red[256];
    const int row = blockIdx.x, tid = threadIdx.x;
    const float* pa = a + (size_t)row * D;
    const float* pr = r + (size_t)row * D;
    const float x0 = pa[tid] + pr[tid];
    const float x1 = pa[tid + 256] + pr[tid + 256];
    red[tid] = x0 + x1;
    __syncthreads();
    for (int off = 128; off > 0; off >>= 1) {
        if (tid < off) red[tid] += red[tid + off];
        __syncthreads();
    }
    const float mean = red[0] * (1.0f / D);
    __syncthreads();
    const float d0 = x0 - mean, d1 = x1 - mean;
    red[tid] = d0 * d0 + d1 * d1;
    __syncthreads();
    for (int off = 128; off > 0; off >>= 1) {
        if (tid < off) red[tid] += red[tid + off];
        __syncthreads();
    }
    const float inv = rsqrtf(red[0] * (1.0f / D) + 1e-5f);
    out[(size_t)row * D + tid]       = d0 * inv * g[tid] + be[tid];
    out[(size_t)row * D + tid + 256] = d1 * inv * g[tid + 256] + be[tid + 256];
}

extern "C" void kernel_launch(void* const* d_in, const int* in_sizes, int n_in,
                              void* d_out, int out_size, void* d_ws, size_t ws_size,
                              hipStream_t stream) {
    (void)in_sizes; (void)n_in; (void)out_size; (void)ws_size;
    const float* queries  = (const float*)d_in[0];
    const float* memory   = (const float*)d_in[1];
    const float* mem_pos  = (const float*)d_in[2];
    const float* qry_pos  = (const float*)d_in[3];
    const float* sa_in_w  = (const float*)d_in[4];
    const float* sa_in_b  = (const float*)d_in[5];
    const float* sa_out_w = (const float*)d_in[6];
    const float* sa_out_b = (const float*)d_in[7];
    const float* ln1_g    = (const float*)d_in[8];
    const float* ln1_b    = (const float*)d_in[9];
    const float* q_w      = (const float*)d_in[10];
    const float* q_b      = (const float*)d_in[11];
    const float* k_w      = (const float*)d_in[12];
    const float* k_b      = (const float*)d_in[13];
    const float* v_w      = (const float*)d_in[14];
    const float* v_b      = (const float*)d_in[15];
    const float* o_w      = (const float*)d_in[16];
    const float* o_b      = (const float*)d_in[17];
    const float* ln2_g    = (const float*)d_in[18];
    const float* ln2_b    = (const float*)d_in[19];
    const float* beta     = (const float*)d_in[20];
    const float* dk_w1    = (const float*)d_in[21];
    const float* dk_b1    = (const float*)d_in[22];
    const float* dk_w2    = (const float*)d_in[23];
    const float* dk_b2    = (const float*)d_in[24];
    const float* ffn_w1   = (const float*)d_in[25];
    const float* ffn_b1   = (const float*)d_in[26];
    const float* ffn_w2   = (const float*)d_in[27];
    const float* ffn_b2   = (const float*)d_in[28];
    const float* ln3_g    = (const float*)d_in[29];
    const float* ln3_b    = (const float*)d_in[30];
    float* out = (float*)d_out;
    float* ws  = (float*)d_ws;

    constexpr int B = 2, Qn = 512, L = 4096, D = 512, H = 8, DFF = 2048;
    constexpr int NQ = B * Qn, NL = B * L;

    float* qkv  = ws;                               // NQ x 3D
    float* attn = qkv  + (size_t)NQ * 3 * D;        // NQ x D  (reused 3x)
    float* x1   = attn + (size_t)NQ * D;            // NQ x D
    float* qh   = x1   + (size_t)NQ * D;            // NQ x D
    float* kh   = qh   + (size_t)NQ * D;            // NL x D
    float* vh   = kh   + (size_t)NL * D;            // NL x D
    float* bia  = vh   + (size_t)NL * D;            // B*Qn*L
    float* x2   = bia  + (size_t)B * Qn * L;        // NQ x D
    float* ffn  = x2   + (size_t)NQ * D;            // NQ x DFF (+ o-proj temp)
    float* tmp  = ffn;

    const float scale = 0.125f;  // HD^-0.5, HD=64

    // 1. fused self-attention qkv projection
    gemm_nt_kernel<<<dim3((3 * D) / 64, NQ / 64), 128, 0, stream>>>(
        queries, sa_in_w, sa_in_b, qkv, D, D, D, 3 * D, 0);
    // 2. self-attention (streaming softmax, no bias)
    attn_kernel<<<dim3(Qn / 16, H, B), 32, 0, stream>>>(
        qkv, qkv + D, qkv + 2 * D, nullptr, nullptr, attn,
        Qn, Qn, 3 * D, 3 * D, D, scale);
    // 3. self out-projection -> tmp
    gemm_nt_kernel<<<dim3(D / 64, NQ / 64), 128, 0, stream>>>(
        attn, sa_out_w, sa_out_b, tmp, D, D, D, D, 0);
    // 4. x1 = ln1(queries + tmp)
    add_ln_kernel<<<NQ, 256, 0, stream>>>(queries, tmp, ln1_g, ln1_b, x1);
    // 5-7. cross-attention Q/K/V projections
    gemm_nt_kernel<<<dim3(D / 64, NQ / 64), 128, 0, stream>>>(
        x1, q_w, q_b, qh, D, D, D, D, 0);
    gemm_nt_kernel<<<dim3(D / 64, NL / 64), 128, 0, stream>>>(
        memory, k_w, k_b, kh, D, D, D, D, 0);
    gemm_nt_kernel<<<dim3(D / 64, NL / 64), 128, 0, stream>>>(
        memory, v_w, v_b, vh, D, D, D, D, 0);
    // 8. distance-kernel bias (head-independent, computed once)
    const long total = (long)B * Qn * L;
    distbias_kernel<<<(unsigned)((total + 255) / 256), 256, 0, stream>>>(
        qry_pos, mem_pos, dk_w1, dk_b1, dk_w2, dk_b2, bia, Qn, L, total);
    // 9. cross-attention with bias (streaming; avoids 134MB score tensor)
    attn_kernel<<<dim3(Qn / 16, H, B), 32, 0, stream>>>(
        qh, kh, vh, bia, beta, attn, Qn, L, D, D, D, scale);
    // 10. cross out-projection -> tmp
    gemm_nt_kernel<<<dim3(D / 64, NQ / 64), 128, 0, stream>>>(
        attn, o_w, o_b, tmp, D, D, D, D, 0);
    // 11. x2 = ln2(x1 + tmp)
    add_ln_kernel<<<NQ, 256, 0, stream>>>(x1, tmp, ln2_g, ln2_b, x2);
    // 12. ffn hidden = relu(x2 @ W1^T + b1)
    gemm_nt_kernel<<<dim3(DFF / 64, NQ / 64), 128, 0, stream>>>(
        x2, ffn_w1, ffn_b1, ffn, D, D, D, DFF, 1);
    // 13. ffn out -> attn (reuse)
    gemm_nt_kernel<<<dim3(D / 64, NQ / 64), 128, 0, stream>>>(
        ffn, ffn_w2, ffn_b2, attn, DFF, DFF, DFF, D, 0);
    // 14. out = ln3(x2 + ffn_out)
    add_ln_kernel<<<NQ, 256, 0, stream>>>(x2, attn, ln3_g, ln3_b, out);
}
